// LinkPredictorWithContrastiveLearning_53360673686007
// MI455X (gfx1250) — matrix operand order repaired and checked by reading, not compiled, not run on hardware
//
#include <hip/hip_runtime.h>
#include <hip/hip_bf16.h>

typedef __attribute__((ext_vector_type(16))) _Float16 v16h;
typedef __attribute__((ext_vector_type(8)))  _Float16 v8h;
typedef __attribute__((ext_vector_type(4)))  _Float16 v4h;
typedef __attribute__((ext_vector_type(8)))  float    v8f;
typedef __attribute__((ext_vector_type(4)))  float    v4f;

#define DIM      128      // embedding dim per node
#define KTOT     256      // concat dim (2*DIM) = GEMM K
#define HID      256      // hidden dim = GEMM N
#define TILE_E   96       // edges per workgroup (6 M-tiles); 600000 % 96 == 0
#define NMTILES  6
#define THREADS  256      // 8 waves (wave32)
#define ROW_F16  264      // padded LDS row stride in f16 elems (528 B -> conflict-free b128)

// ---------------------------------------------------------------------------
// Pack W1 (KTOT x HID row-major f32) into WMMA B-fragment layout, f16.
// Fragment (nt, ks) covers B[k0..k0+31, n0..n0+15], k0=ks*32, n0=nt*16.
// B layout used here (contiguous k order, wave32): lane l holds column n0+(l&15);
// lanes 0-15 hold K=k0..k0+15, lanes 16-31 hold K=k0+16..k0+31;
// VGPR v holds the f16 pair {K=2v, K=2v+1} (low half = even K).
// Memory layout: frag*1024B + lane*32B  (8 dwords contiguous per lane).
// ---------------------------------------------------------------------------
__global__ void pack_w1_kernel(const float* __restrict__ W1,
                               unsigned* __restrict__ pw) {
    int tid  = blockIdx.x * blockDim.x + threadIdx.x;   // 0..4095
    int frag = tid >> 5;                                // 0..127 = nt*8 + ks
    int lane = tid & 31;
    int nt = frag >> 3;
    int ks = frag & 7;
    int n  = nt * 16 + (lane & 15);
    int kbase = ks * 32 + ((lane & 16) ? 16 : 0);
    unsigned* dst = pw + (size_t)frag * 256 + lane * 8;
#pragma unroll
    for (int v = 0; v < 8; ++v) {
        int k = kbase + 2 * v;
        union { _Float16 h[2]; unsigned u; } c;
        c.h[0] = (_Float16)W1[(size_t)k * HID + n];
        c.h[1] = (_Float16)W1[(size_t)(k + 1) * HID + n];
        dst[v] = c.u;
    }
}

// ---------------------------------------------------------------------------
// Fused gather -> f16 cvt -> GEMM(WMMA, B register-resident) -> bias+ReLU
//   -> dot(W2)+b2 -> sigmoid
// Wave w owns columns n = w*32 .. w*32+31 (N-tiles 2w, 2w+1) and loops over
// all 6 M-tiles of the 96-edge block. B fragments are loaded into registers
// only AFTER the gather phase to keep peak VGPR pressure low.
// ---------------------------------------------------------------------------
__global__ __launch_bounds__(THREADS)
void edge_mlp_kernel(const float* __restrict__ emb,     // [N_NODES, DIM] f32
                     const int*   __restrict__ eidx,    // [E, 2] i32
                     const v16h*  __restrict__ pw,      // packed W1 fragments
                     const float* __restrict__ b1,      // [HID]
                     const float* __restrict__ w2,      // [HID]
                     const float* __restrict__ b2,      // [1]
                     float* __restrict__ out,           // [E]
                     int nEdges) {
    __shared__ int   sIdx[2 * TILE_E];
    __shared__ float sLogit[8][TILE_E];
    __shared__ __align__(16) _Float16 sX[TILE_E * ROW_F16];

    const int tid    = threadIdx.x;
    const int wave   = tid >> 5;
    const int lane   = tid & 31;
    const int egBase = blockIdx.x * TILE_E;

    // per-lane epilogue constants (columns are fixed for this wave)
    const int colA = wave * 32 + (lane & 15);
    const int colB = colA + 16;
    const float b1a = b1[colA], b1b = b1[colB];
    const float w2a = w2[colA], w2b = w2[colB];

    // ---- stage edge indices -------------------------------------------------
    if (tid < 2 * TILE_E) {
        int g = egBase * 2 + tid;
        sIdx[tid] = (g < 2 * nEdges) ? eidx[g] : 0;
    }
    __syncthreads();

    // ---- gather: one wave per 512B half-row, 16B coalesced per lane --------
    // (low register pressure here: B fragments are not yet live)
#pragma unroll
    for (int it = 0; it < (2 * TILE_E) / 8; ++it) {
        int hr    = it * 8 + wave;          // half-row 0..191
        int e     = hr >> 1;
        int which = hr & 1;                 // 0 = src emb, 1 = dst emb
        int node  = sIdx[2 * e + which];
        v4f f = *((const v4f*)(emb + (size_t)node * DIM) + lane);
        v4h h;
        h.x = (_Float16)f.x; h.y = (_Float16)f.y;
        h.z = (_Float16)f.z; h.w = (_Float16)f.w;
        *(v4h*)(&sX[e * ROW_F16 + which * DIM + lane * 4]) = h;
    }
    __syncthreads();

    // ---- B fragments: register-resident for this wave's 2 N-tiles ----------
    // Fragments (wave*16 + j), j = 0..7 -> nt = 2*wave, j = 8..15 -> 2*wave+1.
    v16h bfrag[16];
#pragma unroll
    for (int j = 0; j < 16; ++j)
        bfrag[j] = pw[(size_t)(wave * 16 + j) * 32 + lane];

    // A fragment addressing: lane l holds row m = mt*16 + (l&15);
    // lanes 0-15: K 0..15 ; lanes 16-31: K 16..31.
    const int rloc = lane & 15;
    const int koff = (lane & 16) ? 16 : 0;

    // ---- GEMM over 6 M-tiles, B fragments from registers -------------------
#pragma unroll
    for (int mt = 0; mt < NMTILES; ++mt) {
        const _Float16* abase = &sX[(mt * 16 + rloc) * ROW_F16 + koff];
        v8f ca = {};   // accumulator for nt = 2*wave
        v8f cb = {};   // accumulator for nt = 2*wave + 1
#pragma unroll
        for (int ks = 0; ks < 8; ++ks) {
            union { v16h v; v8h h[2]; } a;
            a.h[0] = *(const v8h*)(abase + ks * 32);        // K = k0+koff   ..
            a.h[1] = *(const v8h*)(abase + ks * 32 + 8);    // K = k0+8+koff..
            ca = __builtin_amdgcn_wmma_f32_16x16x32_f16(
                     false, a.v, false, bfrag[ks],     (short)0, ca, false, false);
            cb = __builtin_amdgcn_wmma_f32_16x16x32_f16(
                     false, a.v, false, bfrag[8 + ks], (short)0, cb, false, false);
        }

        // epilogue: bias + ReLU + second-layer partial dot over this wave's
        // 32 columns, then butterfly-reduce the 16 lanes of each half.
        float lg[8];
#pragma unroll
        for (int r = 0; r < 8; ++r) {
            float ha = ca[r] + b1a; ha = ha > 0.0f ? ha : 0.0f;
            float hb = cb[r] + b1b; hb = hb > 0.0f ? hb : 0.0f;
            lg[r] = ha * w2a + hb * w2b;
        }
#pragma unroll
        for (int r = 0; r < 8; ++r) {
            float v = lg[r];
            v += __shfl_xor(v, 1, 32);
            v += __shfl_xor(v, 2, 32);
            v += __shfl_xor(v, 4, 32);
            v += __shfl_xor(v, 8, 32);
            lg[r] = v;
        }
        // C layout: lane holds rows m = mt*16 + (lane>=16 ? 8 : 0) + r
        if ((lane & 15) == 0) {
            int mbase = mt * 16 + ((lane & 16) ? 8 : 0);
#pragma unroll
            for (int r = 0; r < 8; ++r)
                sLogit[wave][mbase + r] = lg[r];
        }
    }
    __syncthreads();

    // ---- combine the 8 waves' column-partials, sigmoid, write ---------------
    if (tid < TILE_E) {
        int ge = egBase + tid;
        if (ge < nEdges) {
            float x = b2[0];
#pragma unroll
            for (int w = 0; w < 8; ++w) x += sLogit[w][tid];
            out[ge] = 1.0f / (1.0f + __expf(-x));
        }
    }
}

extern "C" void kernel_launch(void* const* d_in, const int* in_sizes, int n_in,
                              void* d_out, int out_size, void* d_ws, size_t ws_size,
                              hipStream_t stream) {
    const float* emb  = (const float*)d_in[0];   // emd_all   [100000,128] f32
    const int*   eidx = (const int*)  d_in[1];   // edge_index[600000,2]   i32
    const float* W1   = (const float*)d_in[2];   // [256,256] f32
    const float* b1   = (const float*)d_in[3];   // [256]     f32
    const float* W2   = (const float*)d_in[4];   // [256,1]   f32
    const float* b2   = (const float*)d_in[5];   // [1]       f32
    float* out = (float*)d_out;

    const int nEdges = in_sizes[1] / 2;

    // 128 KB of workspace: W1 pre-packed into WMMA B-fragment layout.
    unsigned* pw = (unsigned*)d_ws;
    pack_w1_kernel<<<16, 256, 0, stream>>>(W1, pw);

    const int nBlocks = (nEdges + TILE_E - 1) / TILE_E;
    edge_mlp_kernel<<<nBlocks, THREADS, 0, stream>>>(
        emb, eidx, (const v16h*)pw, b1, W2, b2, out, nEdges);
}
